// Im2grid_61108794687988
// MI455X (gfx1250) — matrix operand order repaired and checked by reading, not compiled, not run on hardware
//
#include <hip/hip_runtime.h>

typedef __attribute__((ext_vector_type(16))) _Float16 v16h;
typedef __attribute__((ext_vector_type(8)))  float    v8f;
typedef unsigned int u32x4 __attribute__((ext_vector_type(4)));
typedef int          i32x4 __attribute__((ext_vector_type(4)));
typedef int          i32x8 __attribute__((ext_vector_type(8)));

#define PI_F 3.14159265358979323846f

#if defined(__has_builtin)
#if __has_builtin(__builtin_amdgcn_tensor_load_to_lds) && \
    __has_builtin(__builtin_amdgcn_s_wait_tensorcnt)
#define USE_TDM 1
#endif
#endif
#ifndef USE_TDM
#define USE_TDM 0
#endif

static __device__ __forceinline__ int clampi(int v, int lo, int hi) {
  return v < lo ? lo : (v > hi ? hi : v);
}

#if USE_TDM
// Issue one TDM 2-D tile load: rows x NP columns of an fp16 [*, NP] row-major
// tensor starting at gptr (tile start), into LDS at byte offset lds_off.
// D# layout per CDNA5 ISA (normal mode, 2-D tile).
static __device__ __forceinline__
void tdm_load_w_tile(unsigned lds_off, const _Float16* gptr, int rows, int Np) {
  unsigned long long ga = (unsigned long long)gptr;
  u32x4 g0;
  g0.x = 1u;                                           // count=1, user mode
  g0.y = lds_off;                                      // lds_addr (bytes)
  g0.z = (unsigned)(ga & 0xffffffffu);                 // global_addr[31:0]
  g0.w = (unsigned)((ga >> 32) & 0x01ffffffu) | (2u << 30);  // addr[56:32]|type=2
  i32x8 g1;
  g1[0] = (1 << 16);                                   // data_size=1 (2 bytes)
  g1[1] = (int)(((unsigned)Np & 0xffffu) << 16);       // tensor_dim0[15:0]
  g1[2] = (int)((((unsigned)Np >> 16) & 0xffffu) |     // tensor_dim0[31:16]
                (((unsigned)rows & 0xffffu) << 16));   // tensor_dim1[15:0]
  g1[3] = (int)((((unsigned)rows >> 16) & 0xffffu) |   // tensor_dim1[31:16]
                (((unsigned)Np & 0xffffu) << 16));     // tile_dim0 = Np
  g1[4] = (int)((unsigned)rows & 0xffffu);             // tile_dim1 = rows
  g1[5] = Np;                                          // tensor_dim0_stride lo
  g1[6] = 0;
  g1[7] = 0;
  i32x4 z4 = {0, 0, 0, 0};
#if __clang_major__ >= 23
  i32x8 z8 = {0, 0, 0, 0, 0, 0, 0, 0};
  __builtin_amdgcn_tensor_load_to_lds(g0, g1, z4, z4, z8, 0);
#else
  __builtin_amdgcn_tensor_load_to_lds(g0, g1, z4, z4, 0);
#endif
}
#endif

// ---------------------------------------------------------------------------
// conv3d 3x3x3 pad=1 as implicit GEMM via WMMA f32 += f16 x f16.
// M = voxels (tile 16), N = all Cout tiles in-wave (NT = NP/16), K = Cin*27
// padded to 32 (Kp). One A fragment feeds NT WMMAs (A-gather amortized).
// A: branchless per-lane im2col gather from fp16 activations using a k-LUT.
// B: fp16 weight slab [Kp][NP], streamed through LDS by the Tensor Data Mover
//    in double-buffered full-width chunks of KC = 8192/NP rows (32 KB LDS).
// ---------------------------------------------------------------------------
template <int NP>
__global__ __launch_bounds__(32)
void conv3d_wmma_kernel(const _Float16* __restrict__ act16,
                        const _Float16* __restrict__ w16,
                        const long long* __restrict__ lut,
                        const float* __restrict__ bias,
                        float* __restrict__ out,
                        int Cout, int D, int H, int W, int Kp, int act)
{
  constexpr int NT = NP / 16;     // N-tiles per wave
  constexpr int KC = 8192 / NP;   // K rows per LDS chunk (multiple of 32)

  const int DHW   = D * H * W;
  const int mbase = blockIdx.x * 16;
  const int lane  = threadIdx.x;  // wave32
  const int lhi   = lane >> 4;
  const int llo   = lane & 15;

  const int m    = mbase + llo;   // A-fragment row voxel for this lane
  const bool mok = (m < DHW);
  int vz = 0, vy = 0, vx = 0;
  if (mok) { vz = m / (H * W); int r = m % (H * W); vy = r / W; vx = r % W; }

  v8f acc[NT];
#pragma unroll
  for (int nt = 0; nt < NT; ++nt)
    acc[nt] = (v8f){0.f, 0.f, 0.f, 0.f, 0.f, 0.f, 0.f, 0.f};

#if USE_TDM
  __shared__ _Float16 bsh[2][KC * NP];
  const unsigned ldsbase = (unsigned)(size_t)(void*)&bsh[0][0];
  const unsigned ldsstep = (unsigned)(KC * NP * sizeof(_Float16));
#endif

  const int nch = (Kp + KC - 1) / KC;
#if USE_TDM
  {
    int r0 = Kp < KC ? Kp : KC;
    tdm_load_w_tile(ldsbase, w16, r0, NP);
  }
#endif

  for (int c = 0; c < nch; ++c) {
    const int kbase = c * KC;
    const int klen  = (Kp - kbase) < KC ? (Kp - kbase) : KC;
#if USE_TDM
    if (c + 1 < nch) {
      int nb = kbase + KC;
      int rn = (Kp - nb) < KC ? (Kp - nb) : KC;
      tdm_load_w_tile(ldsbase + ((unsigned)(c + 1) & 1u) * ldsstep,
                      w16 + (size_t)nb * NP, rn, NP);
      __builtin_amdgcn_s_wait_tensorcnt(1);
    } else {
      __builtin_amdgcn_s_wait_tensorcnt(0);
    }
    const _Float16* __restrict__ bp = &bsh[c & 1][0];
#endif

    for (int kb = 0; kb < klen; kb += 32) {
      const int kg = kbase + kb;
      v16h a;
#pragma unroll
      for (int e = 0; e < 16; ++e) {
        // A layout (16-bit A 16x32): K = e(+8 if e>=8) + 8*(lane>=16)
        int kA = kg + e + ((e >= 8) ? 8 : 0) + lhi * 8;
        long long L = lut[kA];
        int delta = (int)L;
        int code  = (int)(L >> 32);
        int dz = (code & 3) - 1;
        int dy = ((code >> 2) & 3) - 1;
        int dx = ((code >> 4) & 3) - 1;
        bool ok = mok && (code & 0x40) &&
                  (unsigned)(vz + dz) < (unsigned)D &&
                  (unsigned)(vy + dy) < (unsigned)H &&
                  (unsigned)(vx + dx) < (unsigned)W;
        int idx = ok ? (m + delta) : 0;      // clamped -> always safe load
        _Float16 hv = act16[idx];
        a[e] = ok ? hv : (_Float16)0;
      }
#pragma unroll
      for (int nt = 0; nt < NT; ++nt) {
        v16h b;
#pragma unroll
        for (int e = 0; e < 16; ++e) {
          // B layout (16-bit B 32x16): K = e(+8 if e>=8) + 8*(lane/16), N = lane%16
          int kk = kb + e + ((e >= 8) ? 8 : 0) + lhi * 8;
#if USE_TDM
          b[e] = bp[kk * NP + nt * 16 + llo];
#else
          b[e] = w16[(size_t)(kbase + kk) * NP + nt * 16 + llo];
#endif
        }
        acc[nt] = __builtin_amdgcn_wmma_f32_16x16x32_f16(
            false, a, false, b, (short)0, acc[nt], false, false);
      }
    }
  }

  // C/D layout: N = lane%16, M = r + 8*(lane/16)
#pragma unroll
  for (int nt = 0; nt < NT; ++nt) {
    int n = nt * 16 + llo;
    if (n < Cout) {
      float bv = bias[n];
#pragma unroll
      for (int r = 0; r < 8; ++r) {
        int vm = mbase + r + 8 * lhi;
        if (vm < DHW) {
          float v = acc[nt][r] + bv;
          if (act) v = (v >= 0.f) ? v : 0.1f * v;
          out[(size_t)n * DHW + vm] = v;
        }
      }
    }
  }
}

// fp32 -> fp16 activation conversion (one pass per conv input; L2-resident)
__global__ __launch_bounds__(256)
void f32_to_f16_kernel(const float* __restrict__ in, _Float16* __restrict__ outp,
                       int n)
{
  int i = blockIdx.x * blockDim.x + threadIdx.x;
  if (i < n) outp[i] = (_Float16)in[i];
}

// Pack fp32 OIDHW conv weights into padded fp16 [Kp][Np]
__global__ __launch_bounds__(256)
void pack_w_kernel(const float* __restrict__ w, _Float16* __restrict__ w16,
                   int K, int Kp, int Cout, int Np)
{
  int idx = blockIdx.x * blockDim.x + threadIdx.x;
  int total = Kp * Np;
  if (idx >= total) return;
  int k = idx / Np, n = idx - (idx / Np) * Np;
  float v = (k < K && n < Cout) ? w[(size_t)n * K + k] : 0.f;
  w16[idx] = (_Float16)v;
}

// Per-layer k-LUT: lut[k] = { ci*DHW + dz*HW + dy*W + dx , (dz+1)|(dy+1)<<2|(dx+1)<<4|valid<<6 }
__global__ __launch_bounds__(256)
void build_lut_kernel(long long* __restrict__ lut, int K, int Kp,
                      int DHW, int HW, int W)
{
  int k = blockIdx.x * blockDim.x + threadIdx.x;
  if (k >= Kp) return;
  long long entry = 0;
  if (k < K) {
    int ci = k / 27; int rr = k - ci * 27;
    int q9 = rr / 9; int dz = q9 - 1; int r2 = rr - q9 * 9;
    int q3 = r2 / 3; int dy = q3 - 1; int dx = r2 - q3 * 3 - 1;
    int delta = ci * DHW + dz * HW + dy * W + dx;
    int code = (dz + 1) | ((dy + 1) << 2) | ((dx + 1) << 4) | 0x40;
    entry = (unsigned int)delta | ((long long)code << 32);
  }
  lut[k] = entry;
}

// Per-channel mean / rsqrt(var+eps): one block per channel.
__global__ __launch_bounds__(256)
void stats_kernel(const float* __restrict__ x, float* __restrict__ stats,
                  int C, int DHW)
{
  __shared__ float ss[256], sq[256];
  int c = blockIdx.x;
  float s = 0.f, q = 0.f;
  for (int i = threadIdx.x; i < DHW; i += 256) {
    float v = x[(size_t)c * DHW + i];
    s += v; q += v * v;
  }
  ss[threadIdx.x] = s; sq[threadIdx.x] = q;
  __syncthreads();
  for (int st = 128; st > 0; st >>= 1) {
    if ((int)threadIdx.x < st) {
      ss[threadIdx.x] += ss[threadIdx.x + st];
      sq[threadIdx.x] += sq[threadIdx.x + st];
    }
    __syncthreads();
  }
  if (threadIdx.x == 0) {
    float mean = ss[0] / (float)DHW;
    float var  = sq[0] / (float)DHW - mean * mean;
    stats[c]     = mean;
    stats[C + c] = rsqrtf(var + 1e-5f);
  }
}

__global__ __launch_bounds__(256)
void inorm_lrelu_kernel(float* __restrict__ x, const float* __restrict__ stats,
                        int C, int DHW)
{
  int idx = blockIdx.x * blockDim.x + threadIdx.x;
  int total = C * DHW;
  if (idx >= total) return;
  int c = idx / DHW;
  float v = (x[idx] - stats[c]) * stats[C + c];
  x[idx] = (v >= 0.f) ? v : 0.1f * v;
}

__global__ __launch_bounds__(256)
void pool_kernel(const float* __restrict__ in, float* __restrict__ outp,
                 int C, int D, int H, int W)
{
  int Do = D / 2, Ho = H / 2, Wo = W / 2;
  int total = C * Do * Ho * Wo;
  int idx = blockIdx.x * blockDim.x + threadIdx.x;
  if (idx >= total) return;
  int t = idx;
  int x = t % Wo; t /= Wo;
  int y = t % Ho; t /= Ho;
  int z = t % Do; int c = t / Do;
  float s = 0.f;
#pragma unroll
  for (int dz = 0; dz < 2; ++dz)
#pragma unroll
    for (int dy = 0; dy < 2; ++dy)
#pragma unroll
      for (int dx = 0; dx < 2; ++dx)
        s += in[(((size_t)c * D + (2 * z + dz)) * H + (2 * y + dy)) * W +
                (2 * x + dx)];
  outp[idx] = s * 0.125f;
}

// PE projection to 6 dims (+cos/sin positional emb), optionally of a trilinearly
// warped source (fuses _warp into _pe; zero padding like reference).
__global__ __launch_bounds__(128)
void pe_warp_kernel(const float* __restrict__ src, const float* __restrict__ flow,
                    const float* __restrict__ pw, const float* __restrict__ pb,
                    const float* __restrict__ palpha, float* __restrict__ outp,
                    int C, int D, int H, int W)
{
  int v = blockIdx.x * blockDim.x + threadIdx.x;
  int DHW = D * H * W;
  if (v >= DHW) return;
  int z = v / (H * W); int r = v % (H * W); int y = r / W; int x = r % W;
  float acc[6];
#pragma unroll
  for (int o = 0; o < 6; ++o) acc[o] = pb[o];

  if (flow) {
    float fz = (float)z + flow[v];
    float fy = (float)y + flow[(size_t)DHW + v];
    float fx = (float)x + flow[2 * (size_t)DHW + v];
    float z0f = floorf(fz), y0f = floorf(fy), x0f = floorf(fx);
    int z0 = (int)z0f, y0 = (int)y0f, x0 = (int)x0f;
    float wz = fz - z0f, wy = fy - y0f, wx = fx - x0f;
    int idx8[8]; float wt8[8];
    int j = 0;
#pragma unroll
    for (int dz = 0; dz < 2; ++dz) {
      int zi = z0 + dz; bool bz = (zi >= 0 && zi < D);
      int zc = clampi(zi, 0, D - 1); float wzz = dz ? wz : 1.f - wz;
#pragma unroll
      for (int dy = 0; dy < 2; ++dy) {
        int yi = y0 + dy; bool by = (yi >= 0 && yi < H);
        int yc = clampi(yi, 0, H - 1); float wyy = dy ? wy : 1.f - wy;
#pragma unroll
        for (int dx = 0; dx < 2; ++dx) {
          int xi = x0 + dx; bool bx = (xi >= 0 && xi < W);
          int xc = clampi(xi, 0, W - 1); float wxx = dx ? wx : 1.f - wx;
          idx8[j] = (zc * H + yc) * W + xc;
          wt8[j]  = (bz && by && bx) ? wzz * wyy * wxx : 0.f;
          ++j;
        }
      }
    }
    for (int c = 0; c < C; ++c) {
      const float* sc = src + (size_t)c * DHW;
      float val = 0.f;
#pragma unroll
      for (int t = 0; t < 8; ++t) val += sc[idx8[t]] * wt8[t];
#pragma unroll
      for (int o = 0; o < 6; ++o) acc[o] += val * pw[o * C + c];
    }
  } else {
    for (int c = 0; c < C; ++c) {
      float val = src[(size_t)c * DHW + v];
#pragma unroll
      for (int o = 0; o < 6; ++o) acc[o] += val * pw[o * C + c];
    }
  }

  float a  = palpha[0];
  float az = (float)z * (PI_F / (float)(D - 1));
  float ay = (float)y * (PI_F / (float)(H - 1));
  float ax = (float)x * (PI_F / (float)(W - 1));
  acc[0] += a * cosf(az); acc[1] += a * sinf(az);
  acc[2] += a * cosf(ay); acc[3] += a * sinf(ay);
  acc[4] += a * cosf(ax); acc[5] += a * sinf(ax);
#pragma unroll
  for (int o = 0; o < 6; ++o) outp[(size_t)v * 6 + o] = acc[o];
}

// 27-tap windowed attention -> flow offsets (zero logits at borders, per ref pad)
__global__ __launch_bounds__(128)
void cotr_kernel(const float* __restrict__ q, const float* __restrict__ k,
                 float* __restrict__ flow, int D, int H, int W)
{
  int v = blockIdx.x * blockDim.x + threadIdx.x;
  int DHW = D * H * W;
  if (v >= DHW) return;
  int z = v / (H * W); int r = v % (H * W); int y = r / W; int x = r % W;
  float qv[6];
#pragma unroll
  for (int o = 0; o < 6; ++o) qv[o] = q[(size_t)v * 6 + o];

  float logit[27]; float mx = -1e30f;
  int o = 0;
  for (int a = -1; a <= 1; ++a)
    for (int b = -1; b <= 1; ++b)
      for (int c = -1; c <= 1; ++c) {
        int zz = z + a, yy = y + b, xx = x + c;
        float l = 0.f;
        if ((unsigned)zz < (unsigned)D && (unsigned)yy < (unsigned)H &&
            (unsigned)xx < (unsigned)W) {
          const float* kp = k + (size_t)((zz * H + yy) * W + xx) * 6;
#pragma unroll
          for (int i = 0; i < 6; ++i) l += qv[i] * kp[i];
        }
        logit[o] = l; mx = fmaxf(mx, l); ++o;
      }
  float s = 0.f;
#pragma unroll
  for (int i = 0; i < 27; ++i) { logit[i] = expf(logit[i] - mx); s += logit[i]; }
  float inv = 1.f / s;
  float fz = 0.f, fy = 0.f, fx = 0.f;
  o = 0;
  for (int a = -1; a <= 1; ++a)
    for (int b = -1; b <= 1; ++b)
      for (int c = -1; c <= 1; ++c) {
        float p = logit[o] * inv;
        fz += p * (float)a; fy += p * (float)b; fx += p * (float)c; ++o;
      }
  flow[v] = fz; flow[(size_t)DHW + v] = fy; flow[2 * (size_t)DHW + v] = fx;
}

// out[c] = trilinear(src at id+flow) (+ addf[c] if given)  -- zero padding
__global__ __launch_bounds__(128)
void warp_kernel(const float* __restrict__ src, const float* __restrict__ flow,
                 const float* __restrict__ addf, float* __restrict__ outp,
                 int C, int D, int H, int W)
{
  int v = blockIdx.x * blockDim.x + threadIdx.x;
  int DHW = D * H * W;
  if (v >= DHW) return;
  int z = v / (H * W); int r = v % (H * W); int y = r / W; int x = r % W;
  float fz = (float)z + flow[v];
  float fy = (float)y + flow[(size_t)DHW + v];
  float fx = (float)x + flow[2 * (size_t)DHW + v];
  float z0f = floorf(fz), y0f = floorf(fy), x0f = floorf(fx);
  int z0 = (int)z0f, y0 = (int)y0f, x0 = (int)x0f;
  float wz = fz - z0f, wy = fy - y0f, wx = fx - x0f;
  int idx8[8]; float wt8[8];
  int j = 0;
#pragma unroll
  for (int dz = 0; dz < 2; ++dz) {
    int zi = z0 + dz; bool bz = (zi >= 0 && zi < D);
    int zc = clampi(zi, 0, D - 1); float wzz = dz ? wz : 1.f - wz;
#pragma unroll
    for (int dy = 0; dy < 2; ++dy) {
      int yi = y0 + dy; bool by = (yi >= 0 && yi < H);
      int yc = clampi(yi, 0, H - 1); float wyy = dy ? wy : 1.f - wy;
#pragma unroll
      for (int dx = 0; dx < 2; ++dx) {
        int xi = x0 + dx; bool bx = (xi >= 0 && xi < W);
        int xc = clampi(xi, 0, W - 1); float wxx = dx ? wx : 1.f - wx;
        idx8[j] = (zc * H + yc) * W + xc;
        wt8[j]  = (bz && by && bx) ? wzz * wyy * wxx : 0.f;
        ++j;
      }
    }
  }
  for (int c = 0; c < C; ++c) {
    const float* sc = src + (size_t)c * DHW;
    float s = 0.f;
#pragma unroll
    for (int t = 0; t < 8; ++t) s += sc[idx8[t]] * wt8[t];
    if (addf) s += addf[(size_t)c * DHW + v];
    outp[(size_t)c * DHW + v] = s;
  }
}

// Trilinear x2 upsample, align_corners=True, with output scale folded in.
__global__ __launch_bounds__(256)
void up2_kernel(const float* __restrict__ in, float* __restrict__ outp,
                int C, int D, int H, int W, float scale)
{
  int Do = 2 * D, Ho = 2 * H, Wo = 2 * W;
  int total = C * Do * Ho * Wo;
  int idx = blockIdx.x * blockDim.x + threadIdx.x;
  if (idx >= total) return;
  int t = idx;
  int xo = t % Wo; t /= Wo;
  int yo = t % Ho; t /= Ho;
  int zo = t % Do; int c = t / Do;
  float cz = (float)zo * ((float)(D - 1) / (float)(Do - 1));
  float cy = (float)yo * ((float)(H - 1) / (float)(Ho - 1));
  float cx = (float)xo * ((float)(W - 1) / (float)(Wo - 1));
  int z0 = (int)floorf(cz); int z1 = z0 + 1 < D ? z0 + 1 : D - 1; float wz = cz - (float)z0;
  int y0 = (int)floorf(cy); int y1 = y0 + 1 < H ? y0 + 1 : H - 1; float wy = cy - (float)y0;
  int x0 = (int)floorf(cx); int x1 = x0 + 1 < W ? x0 + 1 : W - 1; float wx = cx - (float)x0;
  const float* sc = in + (size_t)c * D * H * W;
#define G(zz, yy, xx) sc[((zz) * H + (yy)) * W + (xx)]
  float v =
      (1.f - wz) * ((1.f - wy) * ((1.f - wx) * G(z0, y0, x0) + wx * G(z0, y0, x1)) +
                    wy * ((1.f - wx) * G(z0, y1, x0) + wx * G(z0, y1, x1))) +
      wz * ((1.f - wy) * ((1.f - wx) * G(z1, y0, x0) + wx * G(z1, y0, x1)) +
            wy * ((1.f - wx) * G(z1, y1, x0) + wx * G(z1, y1, x1)));
#undef G
  outp[idx] = scale * v;
}

// ---------------------------------------------------------------------------
extern "C" void kernel_launch(void* const* d_in, const int* in_sizes, int n_in,
                              void* d_out, int out_size, void* d_ws, size_t ws_size,
                              hipStream_t stream)
{
  (void)in_sizes; (void)n_in; (void)out_size; (void)ws_size;

  const int Ds[5] = {80, 40, 20, 10, 5};
  const int Hs[5] = {96, 48, 24, 12, 6};
  const int Wl[5] = {80, 40, 20, 10, 5};
  const int chan[5] = {8, 16, 32, 64, 128};
  const int encCin[11]  = {1, 4, 8, 8, 16, 16, 32, 32, 64, 64, 128};
  const int encCout[11] = {4, 8, 8, 16, 16, 32, 32, 64, 64, 128, 128};
  const int encLvl[11]  = {0, 0, 0, 1, 1, 2, 2, 3, 3, 4, 4};

  const float* moving = (const float*)d_in[0];
  const float* fixedp = (const float*)d_in[1];

  // ---- bump allocator on d_ws (floats, 256B aligned blocks) ----
  float* ws = (float*)d_ws;
  size_t off = 0;
  auto alloc = [&](size_t n) -> float* {
    float* p = ws + off;
    off += (n + 63) & ~(size_t)63;
    return p;
  };

  // persistent encoder features
  float* M[5]; float* F[5];
  for (int l = 0; l < 5; ++l) M[l] = alloc((size_t)chan[l] * Ds[l] * Hs[l] * Wl[l]);
  for (int l = 0; l < 5; ++l) F[l] = alloc((size_t)chan[l] * Ds[l] * Hs[l] * Wl[l]);

  const size_t scratch0 = off;

  // ---- encoder-phase scratch ----
  float* tmp1  = alloc((size_t)8 * 80 * 96 * 80);
  float* tmp2  = alloc((size_t)8 * 80 * 96 * 80);
  float* stats = alloc(256);
  _Float16* act16 = (_Float16*)alloc(((size_t)8 * 80 * 96 * 80 + 1) / 2);

  size_t wOff[11], lOff[11];
  int Kp_[11], Np_[11];
  size_t wtot = 0, ltot = 0;
  for (int i = 0; i < 11; ++i) {
    int K = encCin[i] * 27;
    Kp_[i] = (K + 31) & ~31;
    Np_[i] = (encCout[i] + 15) & ~15;
    wOff[i] = wtot; wtot += (size_t)Kp_[i] * Np_[i];
    lOff[i] = ltot; ltot += (size_t)Kp_[i];
  }
  _Float16* w16 = (_Float16*)alloc((wtot + 1) / 2);
  long long* lut = (long long*)alloc(ltot * 2);

  for (int i = 0; i < 11; ++i) {
    int K = encCin[i] * 27;
    int total = Kp_[i] * Np_[i];
    pack_w_kernel<<<(total + 255) / 256, 256, 0, stream>>>(
        (const float*)d_in[2 + 2 * i], w16 + wOff[i], K, Kp_[i], encCout[i], Np_[i]);
    int lvl = encLvl[i];
    int D = Ds[lvl], H = Hs[lvl], W = Wl[lvl];
    build_lut_kernel<<<(Kp_[i] + 255) / 256, 256, 0, stream>>>(
        lut + lOff[i], K, Kp_[i], D * H * W, H * W, W);
  }

  auto convL = [&](const float* in, float* outp, int layer, int act) {
    int lvl = encLvl[layer];
    int D = Ds[lvl], H = Hs[lvl], W = Wl[lvl];
    int nIn = encCin[layer] * D * H * W;
    f32_to_f16_kernel<<<(nIn + 255) / 256, 256, 0, stream>>>(in, act16, nIn);
    dim3 g((unsigned)((D * H * W + 15) / 16), 1);
    const _Float16* wp = w16 + wOff[layer];
    const long long* lp = lut + lOff[layer];
    const float* bp = (const float*)d_in[3 + 2 * layer];
    switch (Np_[layer]) {
      case 16:
        conv3d_wmma_kernel<16><<<g, 32, 0, stream>>>(
            act16, wp, lp, bp, outp, encCout[layer], D, H, W, Kp_[layer], act);
        break;
      case 32:
        conv3d_wmma_kernel<32><<<g, 32, 0, stream>>>(
            act16, wp, lp, bp, outp, encCout[layer], D, H, W, Kp_[layer], act);
        break;
      case 64:
        conv3d_wmma_kernel<64><<<g, 32, 0, stream>>>(
            act16, wp, lp, bp, outp, encCout[layer], D, H, W, Kp_[layer], act);
        break;
      default:
        conv3d_wmma_kernel<128><<<g, 32, 0, stream>>>(
            act16, wp, lp, bp, outp, encCout[layer], D, H, W, Kp_[layer], act);
        break;
    }
  };
  auto cibL = [&](const float* in, float* outp, int layer) {
    convL(in, outp, layer, 0);
    int lvl = encLvl[layer];
    int DHW = Ds[lvl] * Hs[lvl] * Wl[lvl];
    int C = encCout[layer];
    stats_kernel<<<C, 256, 0, stream>>>(outp, stats, C, DHW);
    int total = C * DHW;
    inorm_lrelu_kernel<<<(total + 255) / 256, 256, 0, stream>>>(outp, stats, C, DHW);
  };
  auto poolL = [&](const float* in, float* outp, int C, int lvl) {
    int D = Ds[lvl], H = Hs[lvl], W = Wl[lvl];
    int total = C * (D / 2) * (H / 2) * (W / 2);
    pool_kernel<<<(total + 255) / 256, 256, 0, stream>>>(in, outp, C, D, H, W);
  };
  auto encoder = [&](const float* x, float** feat) {
    convL(x, tmp1, 0, 1);          // conv + lrelu (no IN)
    cibL(tmp1, tmp2, 1);
    cibL(tmp2, feat[0], 2);
    poolL(feat[0], tmp1, 8, 0);
    cibL(tmp1, tmp2, 3);
    cibL(tmp2, feat[1], 4);
    poolL(feat[1], tmp1, 16, 1);
    cibL(tmp1, tmp2, 5);
    cibL(tmp2, feat[2], 6);
    poolL(feat[2], tmp1, 32, 2);
    cibL(tmp1, tmp2, 7);
    cibL(tmp2, feat[3], 8);
    poolL(feat[3], tmp1, 64, 3);
    cibL(tmp1, tmp2, 9);
    cibL(tmp2, feat[4], 10);
  };
  encoder(moving, M);
  encoder(fixedp, F);

  // ---- decoder-phase scratch (reuses encoder scratch region) ----
  off = scratch0;
  float* qb    = alloc((size_t)6 * 80 * 96 * 80);
  float* kb    = alloc((size_t)6 * 80 * 96 * 80);
  float* wb    = alloc((size_t)3 * 80 * 96 * 80);
  float* flowA = alloc((size_t)3 * 80 * 96 * 80);
  float* flowB = alloc((size_t)3 * 80 * 96 * 80);

  auto peL = [&](const float* src, const float* flow, int lvl, float* outp) {
    int D = Ds[lvl], H = Hs[lvl], W = Wl[lvl];
    int DHW = D * H * W;
    pe_warp_kernel<<<(DHW + 127) / 128, 128, 0, stream>>>(
        src, flow, (const float*)d_in[24 + 3 * lvl], (const float*)d_in[25 + 3 * lvl],
        (const float*)d_in[26 + 3 * lvl], outp, chan[lvl], D, H, W);
  };
  auto cotrL = [&](int lvl, float* outp) {
    int D = Ds[lvl], H = Hs[lvl], W = Wl[lvl];
    int DHW = D * H * W;
    cotr_kernel<<<(DHW + 127) / 128, 128, 0, stream>>>(qb, kb, outp, D, H, W);
  };
  auto warpL = [&](const float* src, const float* flow, const float* addf,
                   float* outp, int C, int lvl) {
    int D = Ds[lvl], H = Hs[lvl], W = Wl[lvl];
    int DHW = D * H * W;
    warp_kernel<<<(DHW + 127) / 128, 128, 0, stream>>>(src, flow, addf, outp, C, D, H, W);
  };
  auto up2L = [&](const float* in, float* outp, int lvl /*input lvl*/, float scale) {
    int D = Ds[lvl], H = Hs[lvl], W = Wl[lvl];
    int total = 3 * 8 * D * H * W;
    up2_kernel<<<(total + 255) / 256, 256, 0, stream>>>(in, outp, 3, D, H, W, scale);
  };

  // level 4
  peL(F[4], nullptr, 4, qb);
  peL(M[4], nullptr, 4, kb);
  cotrL(4, wb);
  up2L(wb, flowA, 4, 2.0f);                 // flow = up2(2*w)

  // levels 3,2,1
  for (int lvl = 3; lvl >= 1; --lvl) {
    peL(F[lvl], nullptr, lvl, qb);
    peL(M[lvl], flowA, lvl, kb);            // pe(warp(M, flow)) fused
    cotrL(lvl, wb);
    warpL(flowA, wb, wb, flowB, 3, lvl);    // warp(flow, w) + w
    up2L(flowB, flowA, lvl, 2.0f);          // up2(2*(...))
  }

  // level 0 (no upsample)
  float* outY = (float*)d_out;
  float* outFlow = outY + (size_t)80 * 96 * 80;
  peL(F[0], nullptr, 0, qb);
  peL(M[0], flowA, 0, kb);
  cotrL(0, wb);
  warpL(flowA, wb, wb, outFlow, 3, 0);      // final flow -> d_out[614400..]
  warpL(moving, outFlow, nullptr, outY, 1, 0);  // y = warp(moving, flow)
}
